// OffsetAttention_3728031612943
// MI455X (gfx1250) — matrix operand order repaired and checked by reading, not compiled, not run on hardware
//
#include <hip/hip_runtime.h>
#include <hip/hip_bf16.h>
#include <math.h>

// ---------------------------------------------------------------------------
// OffsetAttention for MI455X (gfx1250): all GEMMs via v_wmma_f32_16x16x32_bf16
// with async-to-LDS double-buffered staging of the shared B tile.
// ---------------------------------------------------------------------------

typedef __bf16 bf16_t;
typedef __attribute__((ext_vector_type(16))) __bf16 v16bf;
typedef __attribute__((ext_vector_type(8)))  __bf16 v8bf;
typedef __attribute__((ext_vector_type(8)))  float  v8f;

#define NTOK   4096   // N == HIDDEN
#define EMB    512

static __device__ inline void store_out(float*  p, float v) { *p = v; }
static __device__ inline void store_out(bf16_t* p, float v) { *p = (bf16_t)v; }

// gfx1250 async copy: 16 bytes per lane, global -> LDS, tracked by ASYNCcnt.
// (inline asm: portable across both toolchains per the CDNA5 bridge doc)
static __device__ __forceinline__ void async_copy_b128(unsigned lds_byte_addr,
                                                       const void* gaddr) {
  asm volatile("global_load_async_to_lds_b128 %0, %1, off"
               :: "v"(lds_byte_addr), "v"(gaddr)
               : "memory");
}

// -------------------------- f32 -> bf16 convert ----------------------------
__global__ void cvt_f32_bf16_kernel(const float* __restrict__ in,
                                    bf16_t* __restrict__ out, int n) {
  for (int i = blockIdx.x * blockDim.x + threadIdx.x; i < n;
       i += gridDim.x * blockDim.x)
    out[i] = (bf16_t)in[i];
}

// ------------------------------ WMMA GEMM ----------------------------------
// C[M x N] = A[M x K] * Bt[N x K]^T (+ bias_m per row) (+ bias_n per col)
// A, Bt bf16 row-major; contraction dim K contiguous in BOTH operands.
// Block = 8 waves covering 128 rows x one 64-col strip; the 64x32 B tile is
// shared by all waves and staged in LDS via async loads (double-buffered).
// Each wave: 1 A-frag (global) reused by 4 B-frags (LDS) -> 4 WMMA / K-step.
template <typename OutT>
__global__ void wmma_gemm_bf16_kernel(const bf16_t* __restrict__ A, int lda,
                                      const bf16_t* __restrict__ Bt, int ldb,
                                      OutT* __restrict__ C, int ldc,
                                      int n_strips, int K,
                                      const float* __restrict__ bias_m,
                                      const float* __restrict__ bias_n) {
  __shared__ __align__(32) bf16_t Bs[2][64][32];   // 2 x 4KB double buffer

  const int tn   = blockIdx.x % n_strips;          // 64-col strip
  const int bm   = blockIdx.x / n_strips;          // 128-row block
  const int tm   = bm * 8 + (threadIdx.x >> 5);    // this wave's 16-row tile
  const int lane = threadIdx.x & 31;
  const int lmod = lane & 15;
  const int half = lane >> 4;

  // cooperative B-tile fetch: thread t moves 16B (col = t/4, K-chunk = t%4)
  const int brow   = threadIdx.x >> 2;             // 0..63 (column of C)
  const int bchunk = (threadIdx.x & 3) * 8;        // bf16 elems within K=32
  const bf16_t* __restrict__ Bg =
      Bt + (size_t)(tn * 64 + brow) * ldb + bchunk;

  // prologue: K-step 0 into buffer 0
  async_copy_b128((unsigned)(uintptr_t)&Bs[0][brow][bchunk], Bg);

  // A fragment: lane<16 holds K 0..15 of
  // row M=lane; lane>=16 holds K 16..31 of row M=lane-16.
  const bf16_t* __restrict__ Arow = A + (size_t)(tm * 16 + lmod) * lda;

  v8f acc0 = {}, acc1 = {}, acc2 = {}, acc3 = {};

  int s = 0;
  for (int k0 = 0; k0 < K; k0 += 32, s ^= 1) {
    const bool more = (k0 + 32) < K;
    if (more) {   // kick off next tile into the other buffer
      async_copy_b128((unsigned)(uintptr_t)&Bs[s ^ 1][brow][bchunk],
                      Bg + k0 + 32);
      asm volatile("s_wait_asynccnt 0x1" ::: "memory");  // current buf ready
      __builtin_prefetch(Arow + k0 + 32, 0, 1);
    } else {
      asm volatile("s_wait_asynccnt 0x0" ::: "memory");
    }
    __syncthreads();                                     // all waves' B ready

    const v8bf alo = *reinterpret_cast<const v8bf*>(Arow + k0 + 16 * half);
    const v8bf ahi = *reinterpret_cast<const v8bf*>(Arow + k0 + 8 + 16 * half);
    v16bf af;
#pragma unroll
    for (int i = 0; i < 8; ++i) { af[i] = alo[i]; af[8 + i] = ahi[i]; }

    // B fragment: lane<16 holds K 0..15 of col N=lane; lane>=16 K 16..31.
    const v16bf b0 = *reinterpret_cast<const v16bf*>(&Bs[s][ 0 + lmod][16 * half]);
    const v16bf b1 = *reinterpret_cast<const v16bf*>(&Bs[s][16 + lmod][16 * half]);
    const v16bf b2 = *reinterpret_cast<const v16bf*>(&Bs[s][32 + lmod][16 * half]);
    const v16bf b3 = *reinterpret_cast<const v16bf*>(&Bs[s][48 + lmod][16 * half]);

    acc0 = __builtin_amdgcn_wmma_f32_16x16x32_bf16(false, af, false, b0,
                                                   (short)0, acc0, false, false);
    acc1 = __builtin_amdgcn_wmma_f32_16x16x32_bf16(false, af, false, b1,
                                                   (short)0, acc1, false, false);
    acc2 = __builtin_amdgcn_wmma_f32_16x16x32_bf16(false, af, false, b2,
                                                   (short)0, acc2, false, false);
    acc3 = __builtin_amdgcn_wmma_f32_16x16x32_bf16(false, af, false, b3,
                                                   (short)0, acc3, false, false);

    __syncthreads();   // everyone done reading buf s before it is overwritten
  }

  // C layout: VGPR r of lane -> (M = tm*16 + r + 8*half, N = tn*64 + c*16 + lmod)
  const int col0 = tn * 64 + lmod;
#pragma unroll
  for (int r = 0; r < 8; ++r) {
    const int row = tm * 16 + 8 * half + r;
    const float bm2 = bias_m ? bias_m[row] : 0.0f;
    OutT* __restrict__ crow = C + (size_t)row * ldc + col0;
    store_out(crow + 0,  acc0[r] + bm2 + (bias_n ? bias_n[col0 + 0]  : 0.0f));
    store_out(crow + 16, acc1[r] + bm2 + (bias_n ? bias_n[col0 + 16] : 0.0f));
    store_out(crow + 32, acc2[r] + bm2 + (bias_n ? bias_n[col0 + 32] : 0.0f));
    store_out(crow + 48, acc3[r] + bm2 + (bias_n ? bias_n[col0 + 48] : 0.0f));
  }
}

// ---------------- row max + sum(exp) over rows of m (f32) ------------------
__global__ void row_stats_kernel(const float* __restrict__ m,
                                 float* __restrict__ rmax,
                                 float* __restrict__ rsum, int cols) {
  __shared__ float red[256];
  const int row = blockIdx.x;
  const float* __restrict__ p = m + (size_t)row * cols;
  float mx = -3.402823466e38f;
  for (int c = threadIdx.x; c < cols; c += blockDim.x) mx = fmaxf(mx, p[c]);
  red[threadIdx.x] = mx; __syncthreads();
  for (int s = 128; s > 0; s >>= 1) {
    if (threadIdx.x < s)
      red[threadIdx.x] = fmaxf(red[threadIdx.x], red[threadIdx.x + s]);
    __syncthreads();
  }
  mx = red[0]; __syncthreads();
  float sm = 0.0f;
  for (int c = threadIdx.x; c < cols; c += blockDim.x) sm += __expf(p[c] - mx);
  red[threadIdx.x] = sm; __syncthreads();
  for (int s = 128; s > 0; s >>= 1) {
    if (threadIdx.x < s) red[threadIdx.x] += red[threadIdx.x + s];
    __syncthreads();
  }
  if (threadIdx.x == 0) { rmax[row] = mx; rsum[row] = red[0]; }
}

// ------ s[i][j] = exp(m[j][i]-rmax[j])/rsum[j]  (transposing softmax) ------
__global__ void softmax_transpose_kernel(const float* __restrict__ m,
                                         const float* __restrict__ rmax,
                                         const float* __restrict__ rsum,
                                         bf16_t* __restrict__ s, int dim) {
  __shared__ float t[32][33];
  const int i0 = blockIdx.x * 32;   // column of m == row of s
  const int j0 = blockIdx.y * 32;   // row of m    == column of s
  const int tx = threadIdx.x, ty = threadIdx.y;
#pragma unroll
  for (int k = 0; k < 4; ++k) {
    const int j = j0 + ty + 8 * k;
    const float val =
        __expf(m[(size_t)j * dim + i0 + tx] - rmax[j]) * (1.0f / rsum[j]);
    t[ty + 8 * k][tx] = val;
  }
  __syncthreads();
#pragma unroll
  for (int k = 0; k < 4; ++k) {
    const int i = i0 + ty + 8 * k;
    s[(size_t)i * dim + j0 + tx] = (bf16_t)t[tx][ty + 8 * k];
  }
}

// ----------------------- l1[j] = sum_j' s[j][j'] ---------------------------
__global__ void row_sum_bf16_kernel(const bf16_t* __restrict__ s,
                                    float* __restrict__ l1, int cols) {
  __shared__ float red[256];
  const int row = blockIdx.x;
  const bf16_t* __restrict__ p = s + (size_t)row * cols;
  float sm = 0.0f;
  for (int c = threadIdx.x; c < cols; c += blockDim.x) sm += (float)p[c];
  red[threadIdx.x] = sm; __syncthreads();
  for (int s2 = 128; s2 > 0; s2 >>= 1) {
    if (threadIdx.x < s2) red[threadIdx.x] += red[threadIdx.x + s2];
    __syncthreads();
  }
  if (threadIdx.x == 0) l1[row] = red[0];
}

// ----------------- vT'[e][j] = v[j][e] / l1[j]  (bf16) ---------------------
__global__ void make_vt_kernel(const float* __restrict__ v,
                               const float* __restrict__ l1,
                               bf16_t* __restrict__ vt, int rows, int cols) {
  __shared__ float t[32][33];
  const int j0 = blockIdx.x * 32;   // row of v
  const int e0 = blockIdx.y * 32;   // col of v
  const int tx = threadIdx.x, ty = threadIdx.y;
#pragma unroll
  for (int k = 0; k < 4; ++k)
    t[ty + 8 * k][tx] = v[(size_t)(j0 + ty + 8 * k) * cols + e0 + tx];
  __syncthreads();
  const float inv = 1.0f / l1[j0 + tx];
#pragma unroll
  for (int k = 0; k < 4; ++k)
    vt[(size_t)(e0 + ty + 8 * k) * rows + j0 + tx] =
        (bf16_t)(t[tx][ty + 8 * k] * inv);
}

// ---------------------------------------------------------------------------
extern "C" void kernel_launch(void* const* d_in, const int* in_sizes, int n_in,
                              void* d_out, int out_size, void* d_ws,
                              size_t ws_size, hipStream_t stream) {
  const float* x   = (const float*)d_in[0];   // (4096, 512)
  const float* w_q = (const float*)d_in[1];   // (4096, 512)
  const float* b_q = (const float*)d_in[2];   // (4096,)
  const float* w_k = (const float*)d_in[3];   // (4096, 512)
  const float* b_k = (const float*)d_in[4];   // (4096,)
  const float* w_v = (const float*)d_in[5];   // (512, 512)
  const float* b_v = (const float*)d_in[6];   // (512,)
  float* out = (float*)d_out;                 // (4096, 512)

  // -------------------- workspace carve-up (~185 MB) -----------------------
  char* p = (char*)d_ws;
  auto take = [&](size_t bytes) {
    char* r = p;
    p += (bytes + 255) & ~(size_t)255;
    return r;
  };
  bf16_t* xbf  = (bf16_t*)take((size_t)NTOK * EMB  * 2);  //  4 MB
  bf16_t* wqbf = (bf16_t*)take((size_t)NTOK * EMB  * 2);  //  4 MB
  bf16_t* wkbf = (bf16_t*)take((size_t)NTOK * EMB  * 2);  //  4 MB
  bf16_t* wvbf = (bf16_t*)take((size_t)EMB  * EMB  * 2);  //  0.5 MB
  bf16_t* qT   = (bf16_t*)take((size_t)NTOK * NTOK * 2);  // 32 MB  qT[h][n]
  bf16_t* kT   = (bf16_t*)take((size_t)NTOK * NTOK * 2);  // 32 MB  kT[h][n]
  float*  vmat = (float*) take((size_t)NTOK * EMB  * 4);  //  8 MB  v[n][e]
  float*  mmat = (float*) take((size_t)NTOK * NTOK * 4);  // 64 MB  m[j][i]=a[i][j]
  bf16_t* sbf  = (bf16_t*)take((size_t)NTOK * NTOK * 2);  // 32 MB  s[i][j]
  bf16_t* vt   = (bf16_t*)take((size_t)EMB  * NTOK * 2);  //  4 MB  vT'[e][j]
  float*  rmax = (float*) take((size_t)NTOK * 4);
  float*  rsum = (float*) take((size_t)NTOK * 4);
  float*  l1   = (float*) take((size_t)NTOK * 4);
  (void)ws_size; (void)n_in; (void)in_sizes; (void)out_size;

  // ------------------------- stage 0: f32 -> bf16 --------------------------
  cvt_f32_bf16_kernel<<<2048, 256, 0, stream>>>(x,   xbf,  NTOK * EMB);
  cvt_f32_bf16_kernel<<<2048, 256, 0, stream>>>(w_q, wqbf, NTOK * EMB);
  cvt_f32_bf16_kernel<<<2048, 256, 0, stream>>>(w_k, wkbf, NTOK * EMB);
  cvt_f32_bf16_kernel<<<512,  256, 0, stream>>>(w_v, wvbf, EMB * EMB);

  // grid = (m_tiles/8) * n_strips blocks; 256 threads = 8 waves / block
  auto blocks = [](int m_tiles, int n_strips) {
    return (m_tiles / 8) * n_strips;
  };

  // stage 1: qT[h][n] = wq[h]·x[n] + bq[h] ; kT likewise  (M=4096,N=4096,K=512)
  wmma_gemm_bf16_kernel<bf16_t><<<blocks(256, 64), 256, 0, stream>>>(
      wqbf, EMB, xbf, EMB, qT, NTOK, 64, EMB, b_q, nullptr);
  wmma_gemm_bf16_kernel<bf16_t><<<blocks(256, 64), 256, 0, stream>>>(
      wkbf, EMB, xbf, EMB, kT, NTOK, 64, EMB, b_k, nullptr);
  // stage 1c: v[n][e] = x[n]·wv[e] + bv[e]   (M=4096,N=512,K=512)
  wmma_gemm_bf16_kernel<float><<<blocks(256, 8), 256, 0, stream>>>(
      xbf, EMB, wvbf, EMB, vmat, EMB, 8, EMB, nullptr, b_v);

  // stage 2: m[j][i] = sum_n qT[j][n]*kT[i][n]  (= a^T)   (4096,4096,K=4096)
  wmma_gemm_bf16_kernel<float><<<blocks(256, 64), 256, 0, stream>>>(
      qT, NTOK, kT, NTOK, mmat, NTOK, 64, NTOK, nullptr, nullptr);

  // stage 3: column softmax of a == row softmax of m, write s[i][j] (bf16)
  row_stats_kernel<<<NTOK, 256, 0, stream>>>(mmat, rmax, rsum, NTOK);
  softmax_transpose_kernel<<<dim3(NTOK / 32, NTOK / 32), dim3(32, 8), 0,
                             stream>>>(mmat, rmax, rsum, sbf, NTOK);

  // stage 4: l1[j] = row-sum of s ; vT'[e][j] = v[j][e]/l1[j]
  row_sum_bf16_kernel<<<NTOK, 256, 0, stream>>>(sbf, l1, NTOK);
  make_vt_kernel<<<dim3(NTOK / 32, EMB / 32), dim3(32, 8), 0, stream>>>(
      vmat, l1, vt, NTOK, EMB);

  // stage 5: out[i][e] = sum_j s[i][j] * vT'[e][j]   (M=4096,N=512,K=4096)
  wmma_gemm_bf16_kernel<float><<<blocks(256, 8), 256, 0, stream>>>(
      sbf, NTOK, vt, NTOK, out, EMB, 8, NTOK, nullptr, nullptr);
}
